// RNN_57715770524283
// MI455X (gfx1250) — matrix operand order repaired and checked by reading, not compiled, not run on hardware
//
#include <hip/hip_runtime.h>
#include <hip/hip_bf16.h>
#include <math.h>

// ---------------------------------------------------------------------------
// RNN LM: emb gather -> pre = x@Wxh+b -> scan h=tanh(pre+h@Whh) -> hs@Wlm+b_lm
// All GEMMs on v_wmma_f32_16x16x32_bf16 (f32 accumulate), copy-free ping-pong
// software pipelines. LDS staging: TDM (tensor_load_to_lds, TENSORcnt) for the
// big GEMM A-stripes; GLOBAL_LOAD_ASYNC_TO_LDS_B128 (ASYNCcnt) in the scan.
// ---------------------------------------------------------------------------

#define VOCAB 32000
#define EMB   1024
#define BB    8
#define TT    512
#define MROWS (BB * TT)   // 4096, row = t*8 + b

typedef __attribute__((ext_vector_type(16))) __bf16 v16bf;
typedef __attribute__((ext_vector_type(8)))  float  v8f;

union V16BF { v16bf v; uint4 q[2]; };

// ---- CDNA5 async / TDM support --------------------------------------------
#if defined(__has_builtin)
# if __has_builtin(__builtin_amdgcn_global_load_async_to_lds_b128)
#  define HAS_ASYNC_B128 1
# endif
# if __has_builtin(__builtin_amdgcn_s_wait_asynccnt)
#  define HAS_WAIT_ASYNC 1
# endif
# if __has_builtin(__builtin_amdgcn_tensor_load_to_lds)
#  define HAS_TDM 1
# endif
# if __has_builtin(__builtin_amdgcn_s_wait_tensorcnt)
#  define HAS_WAIT_TENSOR 1
# endif
#endif

#if defined(HAS_ASYNC_B128)
typedef int v4i_ __attribute__((vector_size(16)));
typedef __attribute__((address_space(1))) v4i_ as1_v4i;
typedef __attribute__((address_space(3))) v4i_ as3_v4i;
#endif

__device__ __forceinline__ void wait_async0() {
#if defined(HAS_WAIT_ASYNC)
  __builtin_amdgcn_s_wait_asynccnt(0);
#elif defined(HAS_ASYNC_B128)
  asm volatile("s_wait_asynccnt 0x0" ::: "memory");
#endif
}

// Copy n uint4 chunks global->LDS (each thread strides). Caller barriers.
__device__ __forceinline__ void stripe_to_lds(uint4* lds, const uint4* src,
                                              int n, int tid, int stride) {
#if defined(HAS_ASYNC_B128)
  for (int i = tid; i < n; i += stride)
    __builtin_amdgcn_global_load_async_to_lds_b128(
        (as1_v4i*)(void*)(src + i), (as3_v4i*)(void*)(lds + i), 0, 0);
  wait_async0();
#else
  for (int i = tid; i < n; i += stride) lds[i] = src[i];
#endif
}

#if defined(HAS_TDM)
typedef __attribute__((address_space(3))) void as3v;
typedef unsigned v4u_t __attribute__((ext_vector_type(4)));
typedef int      v8i_t __attribute__((ext_vector_type(8)));
typedef int      v4s_t __attribute__((ext_vector_type(4)));

__device__ __forceinline__ unsigned lds_byte_addr(void* p) {
  return (unsigned)(unsigned long)(as3v*)p;
}

__device__ __forceinline__ void wait_tensor0() {
#if defined(HAS_WAIT_TENSOR)
  __builtin_amdgcn_s_wait_tensorcnt(0);
#else
  asm volatile("s_wait_tensorcnt 0x0" ::: "memory");
#endif
}

// TDM 1-D copy: n8 elements of 8 bytes, global -> LDS.  D# per ISA 08 §8.3/8.4:
// group0 = {count=1 | lds_addr | global_addr | type=2}; group1 = {data_size=8B,
// tensor_dim0=n8, tensor_dim1=1, tile_dim0=n8 (<=65535), tile_dim1=1,
// tensor_dim0_stride=n8, tensor_dim1_stride=n8}.  One issue per wave.
__device__ __forceinline__ void tdm_copy_8b(const void* g, unsigned lds_addr,
                                            unsigned n8) {
  unsigned long ga = (unsigned long)g;
  v4u_t g0;
  g0[0] = 1u;                                   // count=1, user descriptor
  g0[1] = lds_addr;                             // LDS byte address
  g0[2] = (unsigned)ga;                         // global addr [31:0]
  g0[3] = (unsigned)((ga >> 32) & 0x01FFFFFFu)  // global addr [56:32]
        | 0x80000000u;                          // type=2 ("image")
  v8i_t g1;
  g1[0] = 0x00030000;                           // mask=0, data_size=3 (8B)
  g1[1] = (int)((n8 & 0xFFFFu) << 16);          // tensor_dim0[15:0]
  g1[2] = (int)((n8 >> 16) | (1u << 16));       // dim0[31:16] | tensor_dim1=1
  g1[3] = (int)((n8 & 0xFFFFu) << 16);          // dim1 hi=0 | tile_dim0=n8
  g1[4] = 1;                                    // tile_dim1=1, tile_dim2=0
  g1[5] = (int)n8;                              // tensor_dim0_stride[31:0]
  g1[6] = (int)((n8 & 0xFFFFu) << 16);          // stride0 hi=0 | stride1 lo
  g1[7] = (int)(n8 >> 16);                      // stride1 hi
#pragma unroll
  for (int i = 0; i < 4; ++i) g0[i] = __builtin_amdgcn_readfirstlane(g0[i]);
#pragma unroll
  for (int i = 0; i < 8; ++i) g1[i] = __builtin_amdgcn_readfirstlane(g1[i]);
  v4s_t z = (v4s_t)0;
#if __clang_major__ >= 23
  v8i_t z8 = (v8i_t)0;
  __builtin_amdgcn_tensor_load_to_lds(g0, g1, z, z, z8, 0);
#else
  __builtin_amdgcn_tensor_load_to_lds(g0, g1, z, z, 0);
#endif
}
#endif  // HAS_TDM

// Stage nbytes (multiple of 64) global->LDS; 8 waves each DMA 1/8 via TDM,
// falling back to per-lane async b128 copies. Caller barriers afterwards.
__device__ __forceinline__ void stage_stripe(unsigned short* lds,
                                             const unsigned short* src,
                                             int nbytes, int tid) {
#if defined(HAS_TDM)
  int wave = tid >> 5;
  int per  = (nbytes >> 3) / 8;                 // 8-byte elems per wave
  const char* g = (const char*)src + (long)wave * per * 8;
  unsigned la = lds_byte_addr((void*)lds) + (unsigned)(wave * per * 8);
  tdm_copy_8b(g, la, (unsigned)per);
  wait_tensor0();
#else
  stripe_to_lds((uint4*)lds, (const uint4*)src, nbytes / 16, tid, 256);
#endif
}

__device__ __forceinline__ unsigned short f2bf(float f) {
  union { float f; unsigned u; } x; x.f = f;
  unsigned r = x.u + 0x7FFFu + ((x.u >> 16) & 1u);   // round-to-nearest-even
  return (unsigned short)(r >> 16);
}

// ---------------------------------------------------------------------------
// Repack f32 (K x N, row-major) weight into bf16 WMMA B-fragments.
// Tile (kt,nt) = 32x16 block, 1KB contiguous at block index bi = nt*Kt + kt.
// Lane l (l&15 = column n, l>>4 = K half) owns 16 contiguous K values:
// rows kt*32 + (l>>4)*16 + j, j=0..15, stored at P[bi*512 + l*16].
// ---------------------------------------------------------------------------
__global__ void repack_b_kernel(const float* __restrict__ W,
                                unsigned short* __restrict__ P,
                                int K, int N) {
  int  Kt    = K >> 5;
  long Ntl   = N >> 4;
  long total = Ntl * Kt * 32;
  for (long g = (long)blockIdx.x * blockDim.x + threadIdx.x; g < total;
       g += (long)gridDim.x * blockDim.x) {
    int  lane = (int)(g & 31);
    long bi   = g >> 5;
    int  kt   = (int)(bi % Kt);
    long nt   = bi / Kt;
    int  n    = (int)(nt * 16 + (lane & 15));
    int  kb   = kt * 32 + ((lane >> 4) * 16);
    union { unsigned short s[16]; uint4 q[2]; } tmp;
#pragma unroll
    for (int j = 0; j < 16; ++j)
      tmp.s[j] = f2bf(W[(long)(kb + j) * N + n]);
    uint4* dst = (uint4*)(P + bi * 512 + lane * 16);
    dst[0] = tmp.q[0];
    dst[1] = tmp.q[1];
  }
}

// Embedding gather -> bf16 row-major xb[(t*8+b)*EMB + e]
__global__ void embed_kernel(const int* __restrict__ idx,
                             const float* __restrict__ emb,
                             unsigned short* __restrict__ xb) {
  int row = blockIdx.x;            // 0..4095, row = t*8+b
  int t = row >> 3, b = row & 7;
  int tok = idx[b * TT + t];       // idx is (B,T)
  const float4* src = (const float4*)(emb + (long)tok * EMB);
  float4 f = src[threadIdx.x];     // 256 threads * 4 = 1024
  union { unsigned short s[4]; uint2 q; } o;
  o.s[0] = f2bf(f.x); o.s[1] = f2bf(f.y); o.s[2] = f2bf(f.z); o.s[3] = f2bf(f.w);
  *(uint2*)(xb + (long)row * EMB + threadIdx.x * 4) = o.q;
}

// Zero h double-buffers (16x1024 bf16 each, rows 8..15 stay zero = M padding)
// and the grid-barrier counter.
__global__ void init_kernel(unsigned* h0, unsigned* h1, unsigned* cnt) {
  int i = blockIdx.x * blockDim.x + threadIdx.x;
  if (i < 16 * EMB / 2) { h0[i] = 0u; h1[i] = 0u; }
  if (i == 0) *cnt = 0u;
}

// A-fragment (16x32 bf16) from LDS row-major stripe, matching packed B:
// lanes 0-15 hold K {0..15}, lanes 16-31 hold K {16..31}.
__device__ __forceinline__ V16BF load_a_lds(const unsigned short* As,
                                            int row, int koff, int kt) {
  V16BF a;
  const unsigned short* ap = As + row * EMB + kt * 32 + 2 * koff;
  a.q[0] = *(const uint4*)ap;
  a.q[1] = *(const uint4*)(ap + 8);
  return a;
}

__device__ __forceinline__ V16BF load_b_pk(const unsigned short* wp,
                                           long ntile, int Kt, int kt, int lane) {
  V16BF b;
  const unsigned short* bp = wp + (ntile * Kt + kt) * 512 + lane * 16;
  b.q[0] = *(const uint4*)bp;
  b.q[1] = *(const uint4*)(bp + 8);
  return b;
}

#define WMMA_BF16(a, b, c) \
  __builtin_amdgcn_wmma_f32_16x16x32_bf16(false, (a).v, false, (b).v, (short)0, (c), false, false)

// ---------------------------------------------------------------------------
// pre = xb(4096x1024) @ Wxh + b_h (f32 out). Block = 16x512 tile, 8 waves.
// Depth-2 copy-free ping-pong pipeline: even/odd stages own their registers.
// ---------------------------------------------------------------------------
__global__ void __launch_bounds__(256)
gemm_xw_kernel(const unsigned short* __restrict__ xb,
               const unsigned short* __restrict__ wp,
               const float* __restrict__ bias,
               float* __restrict__ out) {
  __shared__ __align__(16) unsigned short As[16 * EMB];
  int mblk = blockIdx.x;                       // 0..255
  int nblk = blockIdx.y;                       // 0..1
  stage_stripe(As, xb + (long)mblk * 16 * EMB, 16 * EMB * 2, threadIdx.x);
  __syncthreads();

  int wave = threadIdx.x >> 5, lane = threadIdx.x & 31;
  int row = lane & 15, koff = (lane >> 4) * 8;
  int n0 = nblk * 32 + wave * 4;               // n-tile base
  const int Kt = EMB / 32;                     // 32, even
  v8f c[4] = {};
  V16BF a_e, a_o, b_e[4], b_o[4];
  a_e = load_a_lds(As, row, koff, 0);
#pragma unroll
  for (int i = 0; i < 4; ++i) b_e[i] = load_b_pk(wp, n0 + i, Kt, 0, lane);
  for (int kt = 0; kt < Kt; kt += 2) {
    a_o = load_a_lds(As, row, koff, kt + 1);   // prefetch odd stage
#pragma unroll
    for (int i = 0; i < 4; ++i) b_o[i] = load_b_pk(wp, n0 + i, Kt, kt + 1, lane);
#pragma unroll
    for (int i = 0; i < 4; ++i) c[i] = WMMA_BF16(a_e, b_e[i], c[i]);
    int k2 = (kt + 2 < Kt) ? kt + 2 : 0;       // prefetch next even stage
    a_e = load_a_lds(As, row, koff, k2);
#pragma unroll
    for (int i = 0; i < 4; ++i) b_e[i] = load_b_pk(wp, n0 + i, Kt, k2, lane);
#pragma unroll
    for (int i = 0; i < 4; ++i) c[i] = WMMA_BF16(a_o, b_o[i], c[i]);
  }
#pragma unroll
  for (int i = 0; i < 4; ++i) {
    int col = (n0 + i) * 16 + (lane & 15);
    float bv = bias[col];
#pragma unroll
    for (int r8 = 0; r8 < 8; ++r8) {
      int r = mblk * 16 + r8 + (lane >> 4) * 8;
      out[(long)r * EMB + col] = c[i][r8] + bv;
    }
  }
}

// ---------------------------------------------------------------------------
// Sequential recurrence: 8 persistent blocks x 8 waves = 64 waves, one 16-col
// slice each. h double-buffered in ws (16x1024 bf16, rows 8..15 zero pad),
// broadcast via LDS each step (per-lane async b128 copies, ASYNCcnt).
// Depth-4 copy-free pipeline (prefetch dist 3) on A (LDS) and B (L2).
// Grid barrier (monotonic counter) between steps.
// ---------------------------------------------------------------------------
__global__ void __launch_bounds__(256)
rnn_kernel(const float* __restrict__ pre,
           const unsigned short* __restrict__ whh,   // packed B
           unsigned short* __restrict__ hbuf0,
           unsigned short* __restrict__ hbuf1,
           unsigned short* __restrict__ hs,          // 4096x1024 bf16 out
           unsigned* __restrict__ cnt, int nblocks) {
  __shared__ __align__(16) unsigned short Hs[16 * EMB];
  int wave = threadIdx.x >> 5, lane = threadIdx.x & 31;
  int ntile = blockIdx.x * 8 + wave;                 // 0..63
  int row = lane & 15, koff = (lane >> 4) * 8;
  const int Kt = EMB / 32;                           // 32, divisible by 4

  for (int t = 0; t < TT; ++t) {
    unsigned short* cur = (t & 1) ? hbuf1 : hbuf0;
    unsigned short* nxt = (t & 1) ? hbuf0 : hbuf1;
    stripe_to_lds((uint4*)Hs, (const uint4*)cur, 16 * EMB / 8, threadIdx.x, 256);
    __syncthreads();

    v8f c = {};
    V16BF a0, a1, a2, a3, b0, b1, b2, b3;
    a0 = load_a_lds(Hs, row, koff, 0); b0 = load_b_pk(whh, ntile, Kt, 0, lane);
    a1 = load_a_lds(Hs, row, koff, 1); b1 = load_b_pk(whh, ntile, Kt, 1, lane);
    a2 = load_a_lds(Hs, row, koff, 2); b2 = load_b_pk(whh, ntile, Kt, 2, lane);
    for (int kt = 0; kt < Kt; kt += 4) {
      int k3 = kt + 3, k4 = kt + 4, k5 = kt + 5, k6 = kt + 6;
      k4 = (k4 < Kt) ? k4 : 0; k5 = (k5 < Kt) ? k5 : 0; k6 = (k6 < Kt) ? k6 : 0;
      a3 = load_a_lds(Hs, row, koff, k3); b3 = load_b_pk(whh, ntile, Kt, k3, lane);
      c = WMMA_BF16(a0, b0, c);
      a0 = load_a_lds(Hs, row, koff, k4); b0 = load_b_pk(whh, ntile, Kt, k4, lane);
      c = WMMA_BF16(a1, b1, c);
      a1 = load_a_lds(Hs, row, koff, k5); b1 = load_b_pk(whh, ntile, Kt, k5, lane);
      c = WMMA_BF16(a2, b2, c);
      a2 = load_a_lds(Hs, row, koff, k6); b2 = load_b_pk(whh, ntile, Kt, k6, lane);
      c = WMMA_BF16(a3, b3, c);
    }
    int col = ntile * 16 + (lane & 15);
#pragma unroll
    for (int r8 = 0; r8 < 8; ++r8) {
      int r = r8 + (lane >> 4) * 8;
      if (r < 8) {                                   // only real batch rows
        float v = tanhf(c[r8] + pre[((long)t * 8 + r) * EMB + col]);
        unsigned short hv = f2bf(v);
        nxt[r * EMB + col] = hv;
        hs[((long)t * 8 + r) * EMB + col] = hv;
      }
    }
    // grid-wide barrier (all 8 blocks co-resident)
    __syncthreads();
    if (threadIdx.x == 0) {
      __threadfence();
      atomicAdd(cnt, 1u);
      unsigned target = (unsigned)nblocks * (unsigned)(t + 1);
      while (atomicAdd(cnt, 0u) < target) { __builtin_amdgcn_s_sleep(2); }
      __threadfence();
    }
    __syncthreads();
  }
}

// ---------------------------------------------------------------------------
// logits = hs(4096x1024) @ Wlm(1024x32000) + b_lm -> out (B,T,V) f32.
// Block = 32 x 640 tile (8 waves x [2 m-tiles x 5 n-tiles]); every B fragment
// feeds 2 wmma. Depth-2 copy-free ping-pong pipeline. Grid 128 x 50.
// Output stores are nontemporal (524MB stream > L2; keep Wlm pack resident).
// ---------------------------------------------------------------------------
__global__ void __launch_bounds__(256)
gemm_lm_kernel(const unsigned short* __restrict__ hs,
               const unsigned short* __restrict__ wp,
               const float* __restrict__ bias,
               float* __restrict__ out) {
  __shared__ __align__(16) unsigned short As[32 * EMB];   // 64KB of 320KB
  int mblk = blockIdx.x;                       // 0..127 (32 rows each)
  int nblk = blockIdx.y;                       // 0..49
  stage_stripe(As, hs + (long)mblk * 32 * EMB, 32 * EMB * 2, threadIdx.x);
  __syncthreads();

  int wave = threadIdx.x >> 5, lane = threadIdx.x & 31;
  int row = lane & 15, koff = (lane >> 4) * 8;
  int n0 = nblk * 40 + wave * 5;               // n-tile base, max 1999
  const int Kt = EMB / 32;
  v8f c[2][5] = {};
  V16BF ae0, ae1, ao0, ao1, b_e[5], b_o[5];
  ae0 = load_a_lds(As, row, koff, 0);
  ae1 = load_a_lds(As + 16 * EMB, row, koff, 0);
#pragma unroll
  for (int i = 0; i < 5; ++i) b_e[i] = load_b_pk(wp, n0 + i, Kt, 0, lane);
  for (int kt = 0; kt < Kt; kt += 2) {
    // prefetch odd stage
    ao0 = load_a_lds(As, row, koff, kt + 1);
    ao1 = load_a_lds(As + 16 * EMB, row, koff, kt + 1);
#pragma unroll
    for (int i = 0; i < 5; ++i) b_o[i] = load_b_pk(wp, n0 + i, Kt, kt + 1, lane);
#pragma unroll
    for (int i = 0; i < 5; ++i) {
      c[0][i] = WMMA_BF16(ae0, b_e[i], c[0][i]);
      c[1][i] = WMMA_BF16(ae1, b_e[i], c[1][i]);
    }
    // prefetch next even stage
    int k2 = (kt + 2 < Kt) ? kt + 2 : 0;
    ae0 = load_a_lds(As, row, koff, k2);
    ae1 = load_a_lds(As + 16 * EMB, row, koff, k2);
#pragma unroll
    for (int i = 0; i < 5; ++i) b_e[i] = load_b_pk(wp, n0 + i, Kt, k2, lane);
#pragma unroll
    for (int i = 0; i < 5; ++i) {
      c[0][i] = WMMA_BF16(ao0, b_o[i], c[0][i]);
      c[1][i] = WMMA_BF16(ao1, b_o[i], c[1][i]);
    }
  }
#pragma unroll
  for (int m2 = 0; m2 < 2; ++m2) {
#pragma unroll
    for (int i = 0; i < 5; ++i) {
      int col = (n0 + i) * 16 + (lane & 15);
      float bv = bias[col];
#pragma unroll
      for (int r8 = 0; r8 < 8; ++r8) {
        int r  = mblk * 32 + m2 * 16 + r8 + (lane >> 4) * 8; // row = t*8+b
        int b_ = r & 7, t_ = r >> 3;
        __builtin_nontemporal_store(c[m2][i][r8] + bv,
            out + ((long)b_ * TT + t_) * (long)VOCAB + col);
      }
    }
  }
}

// ---------------------------------------------------------------------------
extern "C" void kernel_launch(void* const* d_in, const int* in_sizes, int n_in,
                              void* d_out, int out_size, void* d_ws, size_t ws_size,
                              hipStream_t stream) {
  (void)in_sizes; (void)n_in; (void)out_size; (void)ws_size;
  const int*   idx  = (const int*)d_in[0];
  const float* emb  = (const float*)d_in[1];
  const float* Wxh  = (const float*)d_in[2];
  const float* Whh  = (const float*)d_in[3];
  const float* b_h  = (const float*)d_in[4];
  const float* Wlm  = (const float*)d_in[5];
  const float* b_lm = (const float*)d_in[6];
  float* out = (float*)d_out;

  char*  ws  = (char*)d_ws;
  size_t off = 0;
  auto alloc = [&](size_t bytes) -> void* {
    void* p = ws + off;
    off += (bytes + 255) & ~(size_t)255;
    return p;
  };
  unsigned short* wxh_p = (unsigned short*)alloc((size_t)EMB * EMB * 2);
  unsigned short* whh_p = (unsigned short*)alloc((size_t)EMB * EMB * 2);
  unsigned short* wlm_p = (unsigned short*)alloc((size_t)EMB * VOCAB * 2);
  unsigned short* xb    = (unsigned short*)alloc((size_t)MROWS * EMB * 2);
  unsigned short* hsb   = (unsigned short*)alloc((size_t)MROWS * EMB * 2);
  float*          pre   = (float*)alloc((size_t)MROWS * EMB * 4);
  unsigned short* h0    = (unsigned short*)alloc((size_t)16 * EMB * 2);
  unsigned short* h1    = (unsigned short*)alloc((size_t)16 * EMB * 2);
  unsigned*       cnt   = (unsigned*)alloc(256);

  repack_b_kernel<<<256, 256, 0, stream>>>(Wxh, wxh_p, EMB, EMB);
  repack_b_kernel<<<256, 256, 0, stream>>>(Whh, whh_p, EMB, EMB);
  repack_b_kernel<<<2048, 256, 0, stream>>>(Wlm, wlm_p, EMB, VOCAB);
  embed_kernel<<<MROWS, 256, 0, stream>>>(idx, emb, xb);
  init_kernel<<<32, 256, 0, stream>>>((unsigned*)h0, (unsigned*)h1, cnt);
  gemm_xw_kernel<<<dim3(256, 2), 256, 0, stream>>>(xb, wxh_p, b_h, pre);
  rnn_kernel<<<8, 256, 0, stream>>>(pre, whh_p, h0, h1, hsb, cnt, 8);
  gemm_lm_kernel<<<dim3(128, 50), 256, 0, stream>>>(hsb, wlm_p, b_lm, out);
}
